// SelfAttention_24970939858980
// MI455X (gfx1250) — matrix operand order repaired and checked by reading, not compiled, not run on hardware
//
#include <hip/hip_runtime.h>
#include <hip/hip_bf16.h>

// ---------------------------------------------------------------------------
// CDNA5 (gfx1250) bf16 WMMA self-attention pipeline, round 4.
//  - flash tile: 128 rows/block (8 waves), 64-key j-steps, exp2-domain softmax
//  - GEMMs: 128x64 block, BK=64, 32x32 register tile per wave (4 WMMA accs)
//  - async global->LDS staging (GLOBAL_LOAD_ASYNC_TO_LDS_B128): exact arg types
//  - global_prefetch_b8 for next-tile prefetch
// ---------------------------------------------------------------------------

typedef __attribute__((ext_vector_type(16))) __bf16 v16bf;
typedef __attribute__((ext_vector_type(8)))  float  v8f;
typedef int v4i __attribute__((vector_size(16)));  // matches builtin param type

union F16x16 {
    v16bf v;
    uint4 u[2];
};

#if defined(__has_builtin)
#if __has_builtin(__builtin_amdgcn_global_load_async_to_lds_b128) && \
    __has_builtin(__builtin_amdgcn_s_wait_asynccnt)
#define USE_ASYNC_LDS 1
#endif
#endif
#ifndef USE_ASYNC_LDS
#define USE_ASYNC_LDS 0
#endif

// 16-byte global -> LDS copy; async (no VGPR round-trip, ASYNCcnt) if available
__device__ __forceinline__ void stage16(unsigned short* lds,
                                        const unsigned short* g) {
#if USE_ASYNC_LDS
    __builtin_amdgcn_global_load_async_to_lds_b128(
        (__attribute__((address_space(1))) v4i*)(uintptr_t)g,
        (__attribute__((address_space(3))) v4i*)(unsigned)(uintptr_t)lds,
        0, 0);
#else
    *(uint4*)lds = *(const uint4*)g;
#endif
}

__device__ __forceinline__ void stage_wait() {
#if USE_ASYNC_LDS
    __builtin_amdgcn_s_wait_asynccnt(0);
#endif
}

__device__ __forceinline__ v8f wmma_bf16(v16bf a, v16bf b, v8f c) {
    return __builtin_amdgcn_wmma_f32_16x16x32_bf16(
        false, a, false, b, (short)0, c, false, false);
}

__device__ __forceinline__ float fast_exp2(float x) {
#if defined(__has_builtin)
#if __has_builtin(__builtin_amdgcn_exp2f)
    return __builtin_amdgcn_exp2f(x);
#else
    return exp2f(x);
#endif
#else
    return exp2f(x);
#endif
}

// f32 -> bf16 round-to-nearest-even
__device__ __forceinline__ unsigned short f2bf(float f) {
    unsigned int u = __float_as_uint(f);
    u += 0x7FFFu + ((u >> 16) & 1u);
    return (unsigned short)(u >> 16);
}

// A-fragment (16x32 bf16) from LDS [row][k], stride in halfwords.
// lanes 0-15: row=ln, K={0..7,16..23}; lanes 16-31: K={8..15,24..31}.
__device__ __forceinline__ v16bf lds_afrag(const unsigned short* s, int rowBase,
                                           int stride, int kBase) {
    const int lane = threadIdx.x & 31;
    const int ln = lane & 15, hk = lane >> 4;
    const unsigned short* p = s + (rowBase + ln) * stride + kBase + hk * 8;
    F16x16 f;
    f.u[0] = *(const uint4*)(p);
    f.u[1] = *(const uint4*)(p + 16);
    return f.v;
}

// B-fragment (32x16 bf16) from LDS stored [n][k] (column-contiguous over K).
// lanes 0-15 hold K={0..7,16..23}, lanes 16-31 hold K={8..15,24..31}; n = colBase + ln.
__device__ __forceinline__ v16bf lds_bfrag(const unsigned short* s, int colBase,
                                           int stride, int kBase) {
    const int lane = threadIdx.x & 31;
    const int ln = lane & 15, hk = lane >> 4;
    const unsigned short* p = s + (colBase + ln) * stride + kBase + hk * 8;
    F16x16 f;
    f.u[0] = *(const uint4*)(p);
    f.u[1] = *(const uint4*)(p + 16);
    return f.v;
}

__device__ __forceinline__ float rmax16(float v) {
    #pragma unroll
    for (int m = 1; m < 16; m <<= 1) v = fmaxf(v, __shfl_xor(v, m, 16));
    return v;
}
__device__ __forceinline__ float rsum16(float v) {
    #pragma unroll
    for (int m = 1; m < 16; m <<= 1) v += __shfl_xor(v, m, 16);
    return v;
}

// SCALE * log2(e): softmax evaluated with exp2 (bare v_exp_f32)
#define QSCALE_LOG2E 0.18033688011112042f

// ---------------------------------------------------------------------------
__global__ void f2bf_kernel(const float* __restrict__ in,
                            unsigned short* __restrict__ out, int n) {
    int i = blockIdx.x * blockDim.x + threadIdx.x;
    if (i < n) out[i] = f2bf(in[i]);
}

// ---------------------------------------------------------------------------
// QKV projection: X[4096,1024] @ Wqkv[1024,3072] + b -> Q/K/V [2,16,2048,64] bf16
// Block 128x64, BK=64, 8 waves, 32x32 register tile per wave.
// ---------------------------------------------------------------------------
__global__ __launch_bounds__(256) void qkv_gemm(
    const unsigned short* __restrict__ A, const unsigned short* __restrict__ Bw,
    const float* __restrict__ bias, unsigned short* __restrict__ Qb,
    unsigned short* __restrict__ Kb, unsigned short* __restrict__ Vb) {
    __shared__ unsigned short sA[128 * 72];  // [row][k], 64+8 pad
    __shared__ unsigned short sBT[64 * 72];  // [n][k]

    const int tid = threadIdx.x;
    const int lane = tid & 31, w = tid >> 5;
    const int ln = lane & 15, hk = lane >> 4;
    const int n0 = blockIdx.x * 64, m0 = blockIdx.y * 128;
    const int tr = w & 3, tc = w >> 2;  // 4x2 wave grid of 32x32 tiles

    v8f a00 = {}, a01 = {}, a10 = {}, a11 = {};

    for (int k0 = 0; k0 < 1024; k0 += 64) {
        __syncthreads();
        #pragma unroll
        for (int i = 0; i < 4; i++) {  // A tile 128x64 (contiguous -> async)
            int c = tid + 256 * i;
            int r = c >> 3, kc = c & 7;
            stage16(&sA[r * 72 + kc * 8], &A[(m0 + r) * 1024 + k0 + kc * 8]);
            if (kc == 0 && k0 + 64 < 1024)
                __builtin_prefetch(&A[(m0 + r) * 1024 + k0 + 64], 0, 3);
        }
        #pragma unroll
        for (int i = 0; i < 2; i++) {  // B tile 64x64 -> [n][k] (transpose)
            int c = tid + 256 * i;
            int kr = c >> 3, nc = c & 7;
            uint4 q = *(const uint4*)&Bw[(k0 + kr) * 3072 + n0 + nc * 8];
            const unsigned short* e = (const unsigned short*)&q;
            #pragma unroll
            for (int j = 0; j < 8; j++) sBT[(nc * 8 + j) * 72 + kr] = e[j];
            if (nc == 0 && k0 + 64 < 1024)
                __builtin_prefetch(&Bw[(k0 + 64 + kr) * 3072 + n0], 0, 3);
        }
        stage_wait();
        __syncthreads();
        #pragma unroll
        for (int ks = 0; ks < 64; ks += 32) {
            v16bf af0 = lds_afrag(sA, tr * 32, 72, ks);
            v16bf af1 = lds_afrag(sA, tr * 32 + 16, 72, ks);
            v16bf bf0 = lds_bfrag(sBT, tc * 32, 72, ks);
            v16bf bf1 = lds_bfrag(sBT, tc * 32 + 16, 72, ks);
            a00 = wmma_bf16(af0, bf0, a00);
            a01 = wmma_bf16(af0, bf1, a01);
            a10 = wmma_bf16(af1, bf0, a10);
            a11 = wmma_bf16(af1, bf1, a11);
        }
    }

    #pragma unroll
    for (int ri = 0; ri < 2; ri++) {
        #pragma unroll
        for (int v = 0; v < 8; v++) {
            int m = m0 + tr * 32 + ri * 16 + v + 8 * hk;
            int b = m >> 11, n = m & 2047;
            #pragma unroll
            for (int ci = 0; ci < 2; ci++) {
                int nout = n0 + tc * 32 + ci * 16 + ln;
                float val = (ri ? (ci ? a11[v] : a10[v]) : (ci ? a01[v] : a00[v]));
                val += bias[nout];
                int which = nout >> 10, rem = nout & 1023;
                int h = rem >> 6, d = rem & 63;
                if (which == 0) val *= QSCALE_LOG2E;  // scale+log2e folded into Q
                unsigned short* dst = (which == 0) ? Qb : (which == 1) ? Kb : Vb;
                dst[(((b << 4) + h) * 2048 + n) * 64 + d] = f2bf(val);
            }
        }
    }
}

// ---------------------------------------------------------------------------
// Flash attention: block = (b, h, 128-row tile), 256 threads = 8 waves,
// each wave owns 16 rows; 64-key j-steps; softmax in exp2 domain.
// ---------------------------------------------------------------------------
__global__ __launch_bounds__(256) void flash_attn(
    const unsigned short* __restrict__ Qb, const unsigned short* __restrict__ Kb,
    const unsigned short* __restrict__ Vb, unsigned short* __restrict__ Yb) {
    __shared__ unsigned short Qs[128 * 72];  // [row][d]
    __shared__ unsigned short Ks[64 * 72];   // [key][d] -> B-frag for Q@K^T
    __shared__ unsigned short Vs[64 * 72];   // [d][key] -> B-frag for P@V
    __shared__ unsigned short Ps[128 * 72];  // [row][key] -> A-frag P

    const int bid = blockIdx.x;
    const int ib = bid & 15, h = (bid >> 4) & 15, b = bid >> 8;
    const int tid = threadIdx.x;
    const int lane = tid & 31, w = tid >> 5;
    const int ln = lane & 15, hk = lane >> 4;

    const unsigned short* Qg = Qb + (((b << 4) + h) * 2048 + ib * 128) * 64;
    const unsigned short* Kg = Kb + (((b << 4) + h) * 2048) * 64;
    const unsigned short* Vg = Vb + (((b << 4) + h) * 2048) * 64;

    // stage Q block (128x64), contiguous -> async
    #pragma unroll
    for (int i = 0; i < 4; i++) {
        int c = tid + 256 * i;
        int r = c >> 3, dc = c & 7;
        stage16(&Qs[r * 72 + dc * 8], &Qg[r * 64 + dc * 8]);
    }

    const int r0 = w * 16;
    v8f O0 = {}, O1 = {}, O2 = {}, O3 = {};
    float mstat[8], lstat[8];
    #pragma unroll
    for (int v = 0; v < 8; v++) { mstat[v] = -1e30f; lstat[v] = 0.0f; }

    stage_wait();
    __syncthreads();
    const v16bf aq0 = lds_afrag(Qs, r0, 72, 0);
    const v16bf aq1 = lds_afrag(Qs, r0, 72, 32);

    for (int j0 = 0; j0 < 2048; j0 += 64) {
        __syncthreads();
        // K tile 64x64 -> Ks[key][d], contiguous -> async
        #pragma unroll
        for (int i = 0; i < 2; i++) {
            int c = tid + 256 * i;
            int r = c >> 3, dc = c & 7;
            stage16(&Ks[r * 72 + dc * 8], &Kg[(j0 + r) * 64 + dc * 8]);
            if (dc == 0 && j0 + 64 < 2048)
                __builtin_prefetch(&Kg[(j0 + 64 + r) * 64], 0, 3);
        }
        // V tile 64x64 -> Vs[d][key] (transpose on store, VGPR path)
        #pragma unroll
        for (int i = 0; i < 2; i++) {
            int c = tid + 256 * i;
            int r = c >> 3, dc = c & 7;
            uint4 q = *(const uint4*)&Vg[(j0 + r) * 64 + dc * 8];
            const unsigned short* e = (const unsigned short*)&q;
            #pragma unroll
            for (int j = 0; j < 8; j++) Vs[(dc * 8 + j) * 72 + r] = e[j];
            if (dc == 0 && j0 + 64 < 2048)
                __builtin_prefetch(&Vg[(j0 + 64 + r) * 64], 0, 3);
        }
        stage_wait();
        __syncthreads();

        // scores S(16x64): 4 accumulators of 16x16 (exp2-domain, pre-scaled Q)
        v8f S0 = {}, S1 = {}, S2 = {}, S3 = {};
        S0 = wmma_bf16(aq0, lds_bfrag(Ks, 0, 72, 0), S0);
        S0 = wmma_bf16(aq1, lds_bfrag(Ks, 0, 72, 32), S0);
        S1 = wmma_bf16(aq0, lds_bfrag(Ks, 16, 72, 0), S1);
        S1 = wmma_bf16(aq1, lds_bfrag(Ks, 16, 72, 32), S1);
        S2 = wmma_bf16(aq0, lds_bfrag(Ks, 32, 72, 0), S2);
        S2 = wmma_bf16(aq1, lds_bfrag(Ks, 32, 72, 32), S2);
        S3 = wmma_bf16(aq0, lds_bfrag(Ks, 48, 72, 0), S3);
        S3 = wmma_bf16(aq1, lds_bfrag(Ks, 48, 72, 32), S3);

        // online softmax (base 2); row v+8*hk lives in 16 lanes of a half-wave
        #pragma unroll
        for (int v = 0; v < 8; v++) {
            float s0 = S0[v], s1 = S1[v], s2 = S2[v], s3 = S3[v];
            float mx = rmax16(fmaxf(fmaxf(s0, s1), fmaxf(s2, s3)));
            float mnew = fmaxf(mstat[v], mx);
            float p0 = fast_exp2(s0 - mnew);
            float p1 = fast_exp2(s1 - mnew);
            float p2 = fast_exp2(s2 - mnew);
            float p3 = fast_exp2(s3 - mnew);
            float rs = rsum16((p0 + p1) + (p2 + p3));
            float corr = fast_exp2(mstat[v] - mnew);
            lstat[v] = lstat[v] * corr + rs;
            mstat[v] = mnew;
            O0[v] *= corr; O1[v] *= corr; O2[v] *= corr; O3[v] *= corr;
            int row = r0 + v + 8 * hk;
            Ps[row * 72 + ln]      = f2bf(p0);
            Ps[row * 72 + 16 + ln] = f2bf(p1);
            Ps[row * 72 + 32 + ln] = f2bf(p2);
            Ps[row * 72 + 48 + ln] = f2bf(p3);
        }

        // O(16x64) += P(16x64) @ V(64x64)   (Ps rows private to this wave)
        v16bf ap0 = lds_afrag(Ps, r0, 72, 0);
        v16bf ap1 = lds_afrag(Ps, r0, 72, 32);
        O0 = wmma_bf16(ap0, lds_bfrag(Vs, 0, 72, 0), O0);
        O0 = wmma_bf16(ap1, lds_bfrag(Vs, 0, 72, 32), O0);
        O1 = wmma_bf16(ap0, lds_bfrag(Vs, 16, 72, 0), O1);
        O1 = wmma_bf16(ap1, lds_bfrag(Vs, 16, 72, 32), O1);
        O2 = wmma_bf16(ap0, lds_bfrag(Vs, 32, 72, 0), O2);
        O2 = wmma_bf16(ap1, lds_bfrag(Vs, 32, 72, 32), O2);
        O3 = wmma_bf16(ap0, lds_bfrag(Vs, 48, 72, 0), O3);
        O3 = wmma_bf16(ap1, lds_bfrag(Vs, 48, 72, 32), O3);
    }

    // normalize, store attn_out [4096,1024] bf16 (head h at cols h*64..)
    #pragma unroll
    for (int v = 0; v < 8; v++) {
        float inv = 1.0f / lstat[v];
        int n = ib * 128 + r0 + v + 8 * hk;
        unsigned short* dst = Yb + (b * 2048 + n) * 1024 + h * 64;
        dst[0 * 16 + ln] = f2bf(O0[v] * inv);
        dst[1 * 16 + ln] = f2bf(O1[v] * inv);
        dst[2 * 16 + ln] = f2bf(O2[v] * inv);
        dst[3 * 16 + ln] = f2bf(O3[v] * inv);
    }
}

// ---------------------------------------------------------------------------
// Output projection: Y[4096,1024] @ Wproj[1024,1024] + b -> f32 out
// ---------------------------------------------------------------------------
__global__ __launch_bounds__(256) void proj_gemm(
    const unsigned short* __restrict__ A, const unsigned short* __restrict__ Bw,
    const float* __restrict__ bias, float* __restrict__ out) {
    __shared__ unsigned short sA[128 * 72];
    __shared__ unsigned short sBT[64 * 72];

    const int tid = threadIdx.x;
    const int lane = tid & 31, w = tid >> 5;
    const int ln = lane & 15, hk = lane >> 4;
    const int n0 = blockIdx.x * 64, m0 = blockIdx.y * 128;
    const int tr = w & 3, tc = w >> 2;

    v8f a00 = {}, a01 = {}, a10 = {}, a11 = {};

    for (int k0 = 0; k0 < 1024; k0 += 64) {
        __syncthreads();
        #pragma unroll
        for (int i = 0; i < 4; i++) {
            int c = tid + 256 * i;
            int r = c >> 3, kc = c & 7;
            stage16(&sA[r * 72 + kc * 8], &A[(m0 + r) * 1024 + k0 + kc * 8]);
            if (kc == 0 && k0 + 64 < 1024)
                __builtin_prefetch(&A[(m0 + r) * 1024 + k0 + 64], 0, 3);
        }
        #pragma unroll
        for (int i = 0; i < 2; i++) {
            int c = tid + 256 * i;
            int kr = c >> 3, nc = c & 7;
            uint4 q = *(const uint4*)&Bw[(k0 + kr) * 1024 + n0 + nc * 8];
            const unsigned short* e = (const unsigned short*)&q;
            #pragma unroll
            for (int j = 0; j < 8; j++) sBT[(nc * 8 + j) * 72 + kr] = e[j];
            if (nc == 0 && k0 + 64 < 1024)
                __builtin_prefetch(&Bw[(k0 + 64 + kr) * 1024 + n0], 0, 3);
        }
        stage_wait();
        __syncthreads();
        #pragma unroll
        for (int ks = 0; ks < 64; ks += 32) {
            v16bf af0 = lds_afrag(sA, tr * 32, 72, ks);
            v16bf af1 = lds_afrag(sA, tr * 32 + 16, 72, ks);
            v16bf bf0 = lds_bfrag(sBT, tc * 32, 72, ks);
            v16bf bf1 = lds_bfrag(sBT, tc * 32 + 16, 72, ks);
            a00 = wmma_bf16(af0, bf0, a00);
            a01 = wmma_bf16(af0, bf1, a01);
            a10 = wmma_bf16(af1, bf0, a10);
            a11 = wmma_bf16(af1, bf1, a11);
        }
    }

    #pragma unroll
    for (int ri = 0; ri < 2; ri++) {
        #pragma unroll
        for (int v = 0; v < 8; v++) {
            int m = m0 + tr * 32 + ri * 16 + v + 8 * hk;
            #pragma unroll
            for (int ci = 0; ci < 2; ci++) {
                int nout = n0 + tc * 32 + ci * 16 + ln;
                float val = (ri ? (ci ? a11[v] : a10[v]) : (ci ? a01[v] : a00[v]));
                out[m * 1024 + nout] = val + bias[nout];
            }
        }
    }
}

// ---------------------------------------------------------------------------
extern "C" void kernel_launch(void* const* d_in, const int* in_sizes, int n_in,
                              void* d_out, int out_size, void* d_ws, size_t ws_size,
                              hipStream_t stream) {
    const float* x      = (const float*)d_in[0];  // [2,2048,1024]
    const float* w_qkv  = (const float*)d_in[1];  // [1024,3072]
    const float* b_qkv  = (const float*)d_in[2];  // [3072]
    const float* w_proj = (const float*)d_in[3];  // [1024,1024]
    const float* b_proj = (const float*)d_in[4];  // [1024]
    float* out = (float*)d_out;                   // [2,2048,1024]

    char* ws = (char*)d_ws;
    const size_t MB = 1u << 20;
    unsigned short* xb     = (unsigned short*)(ws + 0 * MB);   // 8 MB
    unsigned short* wqkvb  = (unsigned short*)(ws + 8 * MB);   // 6 MB
    unsigned short* wprojb = (unsigned short*)(ws + 14 * MB);  // 2 MB
    unsigned short* Qb     = (unsigned short*)(ws + 16 * MB);  // 8 MB
    unsigned short* Kb     = (unsigned short*)(ws + 24 * MB);  // 8 MB
    unsigned short* Vb     = (unsigned short*)(ws + 32 * MB);  // 8 MB
    unsigned short* Yb     = (unsigned short*)(ws + 40 * MB);  // 8 MB

    f2bf_kernel<<<(4096 * 1024) / 256, 256, 0, stream>>>(x, xb, 4096 * 1024);
    f2bf_kernel<<<(1024 * 3072) / 256, 256, 0, stream>>>(w_qkv, wqkvb, 1024 * 3072);
    f2bf_kernel<<<(1024 * 1024) / 256, 256, 0, stream>>>(w_proj, wprojb, 1024 * 1024);

    qkv_gemm<<<dim3(3072 / 64, 4096 / 128), 256, 0, stream>>>(xb, wqkvb, b_qkv,
                                                              Qb, Kb, Vb);
    flash_attn<<<2 * 16 * (2048 / 128), 256, 0, stream>>>(Qb, Kb, Vb, Yb);
    proj_gemm<<<dim3(1024 / 64, 4096 / 128), 256, 0, stream>>>(Yb, wprojb, b_proj,
                                                               out);
}
